// TemporalGraphWaveNet_22840636080821
// MI455X (gfx1250) — matrix operand order repaired and checked by reading, not compiled, not run on hardware
//
#include <hip/hip_runtime.h>

#define N_NODES 100000
#define N_EDGES 1600000
#define FDIM    128
#define HDIM    128
#define FE_DIM  16
#define NGRAPH  256
#define BN_EPS  1e-5f

typedef __attribute__((ext_vector_type(16))) _Float16 v16h;
typedef __attribute__((ext_vector_type(8)))  _Float16 v8h;
typedef __attribute__((ext_vector_type(8)))  float    v8f;

#define SHUF16(lo, hi) __builtin_shufflevector(lo, hi, 0,1,2,3,4,5,6,7,8,9,10,11,12,13,14,15)

// ---------------------------------------------------------------------------
// Swizzle a weight matrix W[K x 128] (f32, row major) into the WMMA B-fragment
// layout in f16: frag[(nt*4 + kci)*32 + lane][e] = W[kci*32 + (lane>>4)*8 + (e&7) + (e>>3)*16][nt*16 + (lane&15)]
// (zero-padded for k >= K). 32KB per weight; each lane's fragment is 32B contiguous.
// ---------------------------------------------------------------------------
__global__ void __launch_bounds__(256) k_prep_w(const float* __restrict__ W,
                                                _Float16* __restrict__ out, int K) {
  int t = blockIdx.x * blockDim.x + threadIdx.x;   // 0..1023
  if (t >= 1024) return;
  int nt   = t >> 7;
  int kci  = (t >> 5) & 3;
  int lane = t & 31;
  int col  = lane & 15;
  int kb   = kci * 32 + ((lane >> 4) << 3);
  _Float16* o = out + ((size_t)(nt * 4 + kci) * 32 + lane) * 16;
#pragma unroll
  for (int e = 0; e < 16; ++e) {
    int k = kb + (e & 7) + ((e >> 3) << 4);
    o[e] = (k < K) ? (_Float16)W[(size_t)k * 128 + nt * 16 + col] : (_Float16)0.f;
  }
}

// ---------------------------------------------------------------------------
// degrees (dst), edge->graph mapping, per-graph edge counts
// ---------------------------------------------------------------------------
__global__ void k_deg_edge(const int* __restrict__ src, const int* __restrict__ dst,
                           const int* __restrict__ batch_idx,
                           float* __restrict__ deg, int* __restrict__ edge_batch,
                           float* __restrict__ edge_cnt) {
  int e = blockIdx.x * blockDim.x + threadIdx.x;
  if (e >= N_EDGES) return;
  atomicAdd(&deg[dst[e]], 1.0f);
  int g = batch_idx[src[e]];
  edge_batch[e] = g;
  atomicAdd(&edge_cnt[g], 1.0f);
}

// deg -> dinv in place: dinv = rsqrt(deg + 1)   (+1 = self loop)
__global__ void k_dinv(float* __restrict__ d) {
  int n = blockIdx.x * blockDim.x + threadIdx.x;
  if (n < N_NODES) d[n] = rsqrtf(d[n] + 1.0f);
}

// ---------------------------------------------------------------------------
// C[M x 128] = A[M x 128] @ W[128 x 128]   (f16 WMMA, f32 accumulate)
// A tile (16x128) staged once in LDS as f16 by all 256 threads; 8 waves each
// compute one 16x16 tile; B fragments come pre-swizzled (contiguous per lane).
// ---------------------------------------------------------------------------
__global__ void __launch_bounds__(256) k_gemm128(const float* __restrict__ A,
                                                 const _Float16* __restrict__ Wsw,
                                                 float* __restrict__ C) {
  __shared__ _Float16 As[16][136];
  int tid = threadIdx.x;
  int m0  = blockIdx.x * 16;
  {
    int r  = tid >> 4;
    int c0 = (tid & 15) << 3;
    const float4* ap = (const float4*)(A + (size_t)(m0 + r) * 128 + c0);
    float4 f0 = ap[0], f1 = ap[1];
    v8h h;
    h[0] = (_Float16)f0.x; h[1] = (_Float16)f0.y; h[2] = (_Float16)f0.z; h[3] = (_Float16)f0.w;
    h[4] = (_Float16)f1.x; h[5] = (_Float16)f1.y; h[6] = (_Float16)f1.z; h[7] = (_Float16)f1.w;
    *(v8h*)&As[r][c0] = h;
  }
  __syncthreads();

  int lane = tid & 31;
  int wave = tid >> 5;
  int hi   = lane >> 4;
  int row  = lane & 15;
  int col  = lane & 15;
  const v16h* bw = (const v16h*)Wsw + (size_t)wave * 128 + lane;

  v8f acc = {};
#pragma unroll
  for (int kci = 0; kci < 4; ++kci) {
    v8h lo  = *(const v8h*)&As[row][kci * 32 + (hi << 3)];
    v8h hi8 = *(const v8h*)&As[row][kci * 32 + (hi << 3) + 16];
    v16h a  = SHUF16(lo, hi8);
    v16h b  = bw[kci * 32];
    acc = __builtin_amdgcn_wmma_f32_16x16x32_f16(false, a, false, b,
                                                 (short)0, acc, false, false);
  }
#pragma unroll
  for (int v = 0; v < 8; ++v) {
    int r = v + hi * 8;
    C[(size_t)(m0 + r) * 128 + wave * 16 + col] = acc[v];
  }
}

// ---------------------------------------------------------------------------
// agg[dst] += h[src] * dinv[src]*dinv[dst] ; 32 threads/edge, 4 channels each
// ---------------------------------------------------------------------------
__global__ void k_edge_agg(const int* __restrict__ src, const int* __restrict__ dst,
                           const float* __restrict__ h, const float* __restrict__ dinv,
                           float* __restrict__ agg) {
  long long t = (long long)blockIdx.x * blockDim.x + threadIdx.x;
  int e = (int)(t >> 5);
  if (e >= N_EDGES) return;
  int cg = ((int)t & 31) << 2;
  int s = src[e], d = dst[e];
  float nrm = dinv[s] * dinv[d];
  float4 hv = *(const float4*)(h + (size_t)s * 128 + cg);
  float* ap = agg + (size_t)d * 128 + cg;
  atomicAdd(ap + 0, hv.x * nrm);
  atomicAdd(ap + 1, hv.y * nrm);
  atomicAdd(ap + 2, hv.z * nrm);
  atomicAdd(ap + 3, hv.w * nrm);
}

// ---------------------------------------------------------------------------
// pre = agg + h*dinv^2 + bias (in place into agg); per-column sum / sumsq
// blockDim = 128 (one thread per column), 256 rows per block
// ---------------------------------------------------------------------------
__global__ void __launch_bounds__(128) k_self_stats(float* __restrict__ agg,
                                                    const float* __restrict__ h,
                                                    const float* __restrict__ dinv,
                                                    const float* __restrict__ bias,
                                                    float* __restrict__ csum,
                                                    float* __restrict__ csum2) {
  int c  = threadIdx.x;
  int r0 = blockIdx.x * 256;
  int r1 = r0 + 256; if (r1 > N_NODES) r1 = N_NODES;
  float b = bias[c];
  float s = 0.f, s2 = 0.f;
  for (int r = r0; r < r1; ++r) {
    float di = dinv[r];
    size_t i = (size_t)r * 128 + c;
    float o = agg[i] + h[i] * di * di + b;
    agg[i] = o;
    s += o; s2 += o * o;
  }
  atomicAdd(&csum[c], s);
  atomicAdd(&csum2[c], s2);
}

// out = relu((pre - mu) * rsqrt(var+eps) * gamma + beta)
__global__ void k_bn_relu(const float* __restrict__ pre, float* __restrict__ out,
                          const float* __restrict__ csum, const float* __restrict__ csum2,
                          const float* __restrict__ gamma, const float* __restrict__ beta) {
  int i = blockIdx.x * blockDim.x + threadIdx.x;
  if (i >= N_NODES * 128) return;
  int c = i & 127;
  const float invn = 1.0f / (float)N_NODES;
  float mu  = csum[c] * invn;
  float var = csum2[c] * invn - mu * mu;
  float sc  = rsqrtf(var + BN_EPS) * gamma[c];
  float sh  = beta[c] - mu * sc;
  float o = pre[i] * sc + sh;
  out[i] = o > 0.f ? o : 0.f;
}

// ---------------------------------------------------------------------------
// per-graph node sums + counts
// ---------------------------------------------------------------------------
__global__ void k_pool(const float* __restrict__ h, const int* __restrict__ batch_idx,
                       float* __restrict__ gsum, float* __restrict__ gcnt) {
  int i = blockIdx.x * blockDim.x + threadIdx.x;
  if (i >= N_NODES * 128) return;
  int n = i >> 7, c = i & 127;
  int g = batch_idx[n];
  atomicAdd(&gsum[(size_t)g * 128 + c], h[i]);
  if (c == 0) atomicAdd(&gcnt[g], 1.0f);
}

// ---------------------------------------------------------------------------
// fused edge MLP: e = relu(eattr @ We1 + be1) @ We2 + be2, scatter per-graph.
// 256 threads = 8 waves; each wave: 16 edges, hidden staged in LDS as f16.
// Both weight operands come pre-swizzled (contiguous 32B per lane fragment).
// ---------------------------------------------------------------------------
__global__ void __launch_bounds__(256) k_edge_mlp(const float* __restrict__ eattr,
                                                  const _Float16* __restrict__ We1sw,
                                                  const float* __restrict__ be1,
                                                  const _Float16* __restrict__ We2sw,
                                                  const float* __restrict__ be2,
                                                  const int* __restrict__ ebatch,
                                                  float* __restrict__ esum) {
  __shared__ _Float16 hid[8][16][136];
  int lane = threadIdx.x & 31;
  int wave = threadIdx.x >> 5;
  int hi   = lane >> 4;
  int row  = lane & 15;
  int col  = lane & 15;
  int e0   = (blockIdx.x * 8 + wave) * 16;

  const v16h* bw1 = (const v16h*)We1sw + lane;   // + nt*128 (kci = 0 only)
  const v16h* bw2 = (const v16h*)We2sw + lane;   // + (nt*4 + kci)*32

  // ---- layer 1: K = 16 (FE), zero-padded to 32 (padding baked into We1sw) ----
  v16h a;
  {
    const float4* ap = (const float4*)(eattr + (size_t)(e0 + row) * FE_DIM + (hi << 3));
    float4 f0 = ap[0], f1 = ap[1];
    v8h h;
    h[0] = (_Float16)f0.x; h[1] = (_Float16)f0.y; h[2] = (_Float16)f0.z; h[3] = (_Float16)f0.w;
    h[4] = (_Float16)f1.x; h[5] = (_Float16)f1.y; h[6] = (_Float16)f1.z; h[7] = (_Float16)f1.w;
    v8h z = {};
    a = SHUF16(h, z);    // elements 8..15 (K >= 16) are zero
  }
#pragma unroll
  for (int nt = 0; nt < 8; ++nt) {
    v16h b = bw1[nt * 128];
    v8f acc = {};
    acc = __builtin_amdgcn_wmma_f32_16x16x32_f16(false, a, false, b,
                                                 (short)0, acc, false, false);
    float bias = be1[nt * 16 + col];
#pragma unroll
    for (int v = 0; v < 8; ++v) {
      float o = acc[v] + bias;
      hid[wave][v + hi * 8][nt * 16 + col] = (_Float16)(o > 0.f ? o : 0.f);
    }
  }
  __syncthreads();

  // edge -> graph ids for this lane's 8 output rows
  int gv[8];
#pragma unroll
  for (int v = 0; v < 8; ++v) gv[v] = ebatch[e0 + v + hi * 8];

  // ---- layer 2: hidden[16x128] @ We2[128x128] ----
#pragma unroll
  for (int nt = 0; nt < 8; ++nt) {
    v8f acc = {};
#pragma unroll
    for (int kci = 0; kci < 4; ++kci) {
      v8h lo  = *(const v8h*)&hid[wave][row][kci * 32 + (hi << 3)];
      v8h hi8 = *(const v8h*)&hid[wave][row][kci * 32 + (hi << 3) + 16];
      v16h aa = SHUF16(lo, hi8);
      v16h bb = bw2[(nt * 4 + kci) * 32];
      acc = __builtin_amdgcn_wmma_f32_16x16x32_f16(false, aa, false, bb,
                                                   (short)0, acc, false, false);
    }
    float bias = be2[nt * 16 + col];
#pragma unroll
    for (int v = 0; v < 8; ++v)
      atomicAdd(&esum[(size_t)gv[v] * 128 + nt * 16 + col], acc[v] + bias);
  }
}

// ---------------------------------------------------------------------------
// out = gsum/max(gcnt,1) + (ecnt>0 ? esum/max(ecnt,1) : 0)
// ---------------------------------------------------------------------------
__global__ void k_final(const float* __restrict__ gsum, const float* __restrict__ gcnt,
                        const float* __restrict__ esum, const float* __restrict__ ecnt,
                        float* __restrict__ out) {
  int i = blockIdx.x * blockDim.x + threadIdx.x;
  if (i >= NGRAPH * 128) return;
  int g = i >> 7;
  float node = gsum[i] / fmaxf(gcnt[g], 1.0f);
  float edge = ecnt[g] > 0.f ? esum[i] / fmaxf(ecnt[g], 1.0f) : 0.f;
  out[i] = node + edge;
}

// ---------------------------------------------------------------------------
extern "C" void kernel_launch(void* const* d_in, const int* in_sizes, int n_in,
                              void* d_out, int out_size, void* d_ws, size_t ws_size,
                              hipStream_t stream) {
  const float* x      = (const float*)d_in[0];
  const int*   eidx   = (const int*)  d_in[1];
  const int*   src    = eidx;
  const int*   dst    = eidx + N_EDGES;
  const float* eattr  = (const float*)d_in[2];
  const int*   batch  = (const int*)  d_in[3];
  const float* Wg1    = (const float*)d_in[4];
  const float* bg1    = (const float*)d_in[5];
  const float* gamma1 = (const float*)d_in[6];
  const float* beta1  = (const float*)d_in[7];
  const float* Wg2    = (const float*)d_in[8];
  const float* bg2    = (const float*)d_in[9];
  const float* gamma2 = (const float*)d_in[10];
  const float* beta2  = (const float*)d_in[11];
  const float* We1    = (const float*)d_in[12];
  const float* be1    = (const float*)d_in[13];
  const float* We2    = (const float*)d_in[14];
  const float* be2    = (const float*)d_in[15];
  float* out = (float*)d_out;

  // ---- carve workspace (all chunks 32B-aligned) ----
  char* p = (char*)d_ws;
  const size_t NH = (size_t)N_NODES * 128;
  float* hA    = (float*)p; p += NH * sizeof(float);
  float* hB    = (float*)p; p += NH * sizeof(float);
  float* agg   = (float*)p; p += NH * sizeof(float);
  float* dinv  = (float*)p; p += (size_t)N_NODES * sizeof(float);     // deg then dinv
  int*   ebat  = (int*)  p; p += (size_t)N_EDGES * sizeof(int);
  float* csum  = (float*)p; p += 128 * sizeof(float);
  float* csum2 = (float*)p; p += 128 * sizeof(float);
  float* gsum  = (float*)p; p += (size_t)NGRAPH * 128 * sizeof(float);
  float* gcnt  = (float*)p; p += (size_t)NGRAPH * sizeof(float);
  float* esum  = (float*)p; p += (size_t)NGRAPH * 128 * sizeof(float);
  float* ecnt  = (float*)p; p += (size_t)NGRAPH * sizeof(float);
  const size_t WSWELTS = (size_t)8 * 4 * 32 * 16;                     // 16384 halves, 32KB
  _Float16* wswG1 = (_Float16*)p; p += WSWELTS * sizeof(_Float16);
  _Float16* wswG2 = (_Float16*)p; p += WSWELTS * sizeof(_Float16);
  _Float16* wswE1 = (_Float16*)p; p += WSWELTS * sizeof(_Float16);
  _Float16* wswE2 = (_Float16*)p; p += WSWELTS * sizeof(_Float16);

  // ---- zero atomic accumulators (graph-capturable memset nodes) ----
  hipMemsetAsync(dinv, 0, (size_t)N_NODES * sizeof(float), stream);
  hipMemsetAsync(gsum, 0, (size_t)NGRAPH * 128 * sizeof(float), stream);
  hipMemsetAsync(gcnt, 0, (size_t)NGRAPH * sizeof(float), stream);
  hipMemsetAsync(esum, 0, (size_t)NGRAPH * 128 * sizeof(float), stream);
  hipMemsetAsync(ecnt, 0, (size_t)NGRAPH * sizeof(float), stream);

  // ---- pre-swizzle weights into WMMA B-fragment layout (f16) ----
  k_prep_w<<<4, 256, 0, stream>>>(Wg1, wswG1, 128);
  k_prep_w<<<4, 256, 0, stream>>>(Wg2, wswG2, 128);
  k_prep_w<<<4, 256, 0, stream>>>(We1, wswE1, FE_DIM);
  k_prep_w<<<4, 256, 0, stream>>>(We2, wswE2, 128);

  // ---- graph structure ----
  k_deg_edge<<<(N_EDGES + 255) / 256, 256, 0, stream>>>(src, dst, batch, dinv, ebat, ecnt);
  k_dinv<<<(N_NODES + 255) / 256, 256, 0, stream>>>(dinv);

  // ---- GCN layer 1 ----
  hipMemsetAsync(agg, 0, NH * sizeof(float), stream);
  hipMemsetAsync(csum, 0, 2 * 128 * sizeof(float), stream);   // csum + csum2 contiguous
  k_gemm128<<<N_NODES / 16, 256, 0, stream>>>(x, wswG1, hA);
  k_edge_agg<<<(N_EDGES * 32) / 256, 256, 0, stream>>>(src, dst, hA, dinv, agg);
  k_self_stats<<<(N_NODES + 255) / 256, 128, 0, stream>>>(agg, hA, dinv, bg1, csum, csum2);
  k_bn_relu<<<(N_NODES * 128 + 255) / 256, 256, 0, stream>>>(agg, hB, csum, csum2, gamma1, beta1);

  // ---- GCN layer 2 ----
  hipMemsetAsync(agg, 0, NH * sizeof(float), stream);
  hipMemsetAsync(csum, 0, 2 * 128 * sizeof(float), stream);
  k_gemm128<<<N_NODES / 16, 256, 0, stream>>>(hB, wswG2, hA);
  k_edge_agg<<<(N_EDGES * 32) / 256, 256, 0, stream>>>(src, dst, hA, dinv, agg);
  k_self_stats<<<(N_NODES + 255) / 256, 128, 0, stream>>>(agg, hA, dinv, bg2, csum, csum2);
  k_bn_relu<<<(N_NODES * 128 + 255) / 256, 256, 0, stream>>>(agg, hB, csum, csum2, gamma2, beta2);

  // ---- global mean pool ----
  k_pool<<<(N_NODES * 128 + 255) / 256, 256, 0, stream>>>(hB, batch, gsum, gcnt);

  // ---- fused edge MLP + per-graph scatter ----
  k_edge_mlp<<<N_EDGES / 128, 256, 0, stream>>>(eattr, wswE1, be1, wswE2, be2, ebat, esum);

  // ---- combine ----
  k_final<<<(NGRAPH * 128 + 255) / 256, 256, 0, stream>>>(gsum, gcnt, esum, ecnt, out);
}
